// ClusterClassifier_10428180595170
// MI455X (gfx1250) — compile-verified, matrix operand order checked by reading, not run on hardware
//
#include <hip/hip_runtime.h>
#include <stdint.h>

// ---------------------------------------------------------------------------
// Types for CDNA5 WMMA (gfx1250, wave32)
// ---------------------------------------------------------------------------
typedef __attribute__((ext_vector_type(16))) __bf16 v16bf;
typedef __attribute__((ext_vector_type(8)))  float  v8f;

struct alignas(16) U4 { unsigned int x, y, z, w; };

union Frag16 {
  v16bf v;
  unsigned int u[8];
};

__device__ __forceinline__ unsigned short f32_to_bf16(float f) {
  unsigned int u = __float_as_uint(f);
  u = (u + 0x7FFFu + ((u >> 16) & 1u)) >> 16;   // round-to-nearest-even
  return (unsigned short)u;
}

// Load one 16-bit A/B fragment slice for this lane.
// Layout (ISA 7.12.2, 16-bit A 16x32): lane holds row (lane&15),
// K = kb..kb+7 in v0..v3 and K = kb+16..kb+23 in v4..v7, kb=(lane>>4)*8.
// Caller passes (row base pointer, k0 + kb); we fetch 16B + 16B.
__device__ __forceinline__ v16bf load_frag(const unsigned short* __restrict__ base,
                                           int koff) {
  Frag16 f;
  U4 lo = *(const U4*)(base + koff);
  U4 hi = *(const U4*)(base + koff + 16);
  f.u[0] = lo.x; f.u[1] = lo.y; f.u[2] = lo.z; f.u[3] = lo.w;
  f.u[4] = hi.x; f.u[5] = hi.y; f.u[6] = hi.z; f.u[7] = hi.w;
  return f.v;
}

__device__ __forceinline__ v8f wmma_bf16(v16bf a, v16bf b, v8f c) {
  return __builtin_amdgcn_wmma_f32_16x16x32_bf16(
      /*neg_a=*/false, a, /*neg_b=*/false, b,
      /*c_mod=*/(short)0, c, /*reuse_a=*/false, /*reuse_b=*/false);
}

#define HDIM      768
#define NTOK      8192           // B*S = 4*2048
#define NCLUST    8
#define SUMV      36000

// ---------------------------------------------------------------------------
// Kernel: f32 -> bf16 conversion (weights / activations)
// ---------------------------------------------------------------------------
__global__ void convert_bf16_kernel(const float* __restrict__ in,
                                    unsigned short* __restrict__ out, int n) {
  int i = blockIdx.x * blockDim.x + threadIdx.x;
  if (i < n) out[i] = f32_to_bf16(in[i]);
}

// ---------------------------------------------------------------------------
// Kernel: per-cluster token compaction (atomic append).
// Order inside a cluster is nondeterministic but each token scatters its own
// output row, so the final result is deterministic.
// ---------------------------------------------------------------------------
__global__ void compact_kernel(const int* __restrict__ labels,
                               int* __restrict__ counts,
                               int* __restrict__ lists, int ntok) {
  int t = blockIdx.x * blockDim.x + threadIdx.x;
  if (t < ntok) {
    int c = labels[t];
    int pos = atomicAdd(&counts[c], 1);
    lists[c * NTOK + pos] = t;
  }
}

// ---------------------------------------------------------------------------
// Kernel: fused transform  h = LN(GELU(x W^T + b))  -> bf16
// Block: 256 threads (8 waves). Tile: 16 tokens x 768 outputs (full row so
// LayerNorm can be fused). Each wave owns 6 WMMA N-tiles (96 cols).
// ---------------------------------------------------------------------------
__global__ __launch_bounds__(256) void transform_kernel(
    const unsigned short* __restrict__ xbf,   // [8192,768] bf16
    const unsigned short* __restrict__ wbf,   // [768,768]  bf16 (row o, col h)
    const float* __restrict__ bias,
    const float* __restrict__ gamma_,
    const float* __restrict__ beta_,
    unsigned short* __restrict__ hbf)         // [8192,768] bf16
{
  __shared__ float hs[16 * HDIM];             // 48 KB
  const int tid  = threadIdx.x;
  const int wave = tid >> 5;
  const int lane = tid & 31;
  const int lrow = lane & 15;
  const int kb   = (lane >> 4) << 3;
  const int m0   = blockIdx.x << 4;

  v8f acc[6];
#pragma unroll
  for (int t = 0; t < 6; ++t)
    acc[t] = v8f{0.f,0.f,0.f,0.f,0.f,0.f,0.f,0.f};

  const unsigned short* arow = xbf + (size_t)(m0 + lrow) * HDIM;
  const int nb0 = wave * 96;                  // this wave's first column

  for (int k0 = 0; k0 < HDIM; k0 += 32) {
    v16bf a = load_frag(arow, k0 + kb);
#pragma unroll
    for (int t = 0; t < 6; ++t) {
      const unsigned short* brow = wbf + (size_t)(nb0 + t * 16 + lrow) * HDIM;
      v16bf b = load_frag(brow, k0 + kb);
      acc[t] = wmma_bf16(a, b, acc[t]);
    }
  }

  // bias + exact (erf) GELU -> LDS.  C/D layout: m = r + 8*(lane>>4), n = lane&15
  const int mrow = (lane >> 4) << 3;
#pragma unroll
  for (int t = 0; t < 6; ++t) {
    int n = nb0 + t * 16 + lrow;
    float bn = bias[n];
#pragma unroll
    for (int r = 0; r < 8; ++r) {
      float v  = acc[t][r] + bn;
      float ge = 0.5f * v * (1.0f + erff(v * 0.70710678118654752f));
      hs[(mrow + r) * HDIM + n] = ge;
    }
  }
  __syncthreads();

  // LayerNorm: 16 threads per row, butterfly reduce within 16-lane teams.
  const int row = tid >> 4;      // 0..15
  const int cb  = tid & 15;
  float s = 0.f, s2 = 0.f;
#pragma unroll
  for (int j = 0; j < 48; ++j) {
    float v = hs[row * HDIM + cb + j * 16];
    s += v; s2 += v * v;
  }
#pragma unroll
  for (int m = 1; m < 16; m <<= 1) {
    s  += __shfl_xor(s,  m, 32);
    s2 += __shfl_xor(s2, m, 32);
  }
  const float inv = 1.0f / 768.0f;
  float mu  = s * inv;
  float var = s2 * inv - mu * mu;
  float rs  = rsqrtf(var + 1e-12f);

  unsigned short* orow = hbf + (size_t)(m0 + row) * HDIM;
#pragma unroll 4
  for (int j = 0; j < 48; ++j) {
    int c = cb + j * 16;
    float v = hs[row * HDIM + c];
    float o = (v - mu) * rs * gamma_[c] + beta_[c];
    orow[c] = f32_to_bf16(o);
  }
}

// ---------------------------------------------------------------------------
// Kernel: per-cluster decoder GEMM over compacted tokens, scatter store.
// Grid: (vocab tiles of 512, token tiles of 16, 8 clusters). Blocks past the
// live token count or vocab size exit (output pre-zeroed by memset).
// ---------------------------------------------------------------------------
struct DecParams {
  const unsigned short* h;        // [8192,768] bf16
  const unsigned short* wdec;     // [36000,768] bf16, clusters concatenated
  const int* counts;              // [8]
  const int* lists;               // [8][8192]
  const float* biases[NCLUST];
  float* out;
  long long out_off[NCLUST];      // element offset of cluster i in d_out
  int vocab[NCLUST];
  int wrow[NCLUST];               // first row of cluster i inside wdec
};

__global__ __launch_bounds__(256) void decoder_kernel(DecParams p) {
  const int cl  = blockIdx.z;
  const int V   = p.vocab[cl];
  const int cnt = p.counts[cl];
  const int m0  = blockIdx.y << 4;
  if (m0 >= cnt) return;
  const int n0  = blockIdx.x << 9;            // 512 cols per block
  if (n0 >= V) return;

  __shared__ int toks[16];
  const int tid = threadIdx.x;
  if (tid < 16) {
    int mi = m0 + tid;
    if (mi >= cnt) mi = cnt - 1;              // clamp; stores guarded below
    toks[tid] = p.lists[cl * NTOK + mi];
  }
  __syncthreads();

  const int wave = tid >> 5;
  const int lane = tid & 31;
  const int lrow = lane & 15;
  const int kb   = (lane >> 4) << 3;

  const unsigned short* arow  = p.h + (size_t)toks[lrow] * HDIM;
  const unsigned short* wbase = p.wdec + (size_t)p.wrow[cl] * HDIM;

  int nrow[4];
#pragma unroll
  for (int t = 0; t < 4; ++t) {
    int n = n0 + wave * 64 + t * 16 + lrow;
    nrow[t] = (n < V) ? n : (V - 1);          // clamp; stores guarded below
  }

  v8f acc[4];
#pragma unroll
  for (int t = 0; t < 4; ++t)
    acc[t] = v8f{0.f,0.f,0.f,0.f,0.f,0.f,0.f,0.f};

  for (int k0 = 0; k0 < HDIM; k0 += 32) {
    v16bf a = load_frag(arow, k0 + kb);
#pragma unroll
    for (int t = 0; t < 4; ++t) {
      v16bf b = load_frag(wbase + (size_t)nrow[t] * HDIM, k0 + kb);
      acc[t] = wmma_bf16(a, b, acc[t]);
    }
  }

  const float* bias = p.biases[cl];
  float* outc = p.out + p.out_off[cl];
  const int mrow = (lane >> 4) << 3;
#pragma unroll
  for (int t = 0; t < 4; ++t) {
    int n = n0 + wave * 64 + t * 16 + lrow;
    if (n >= V) continue;
    float bn = bias[n];
#pragma unroll
    for (int r = 0; r < 8; ++r) {
      int mi = m0 + mrow + r;
      if (mi < cnt) {
        int tok = toks[mrow + r];
        outc[(size_t)tok * V + n] = acc[t][r] + bn;
      }
    }
  }
}

// ---------------------------------------------------------------------------
// Host launcher
// ---------------------------------------------------------------------------
extern "C" void kernel_launch(void* const* d_in, const int* in_sizes, int n_in,
                              void* d_out, int out_size, void* d_ws, size_t ws_size,
                              hipStream_t stream) {
  (void)in_sizes; (void)n_in; (void)ws_size;

  const float* x      = (const float*)d_in[0];
  const int*   labels = (const int*)  d_in[1];
  const float* W_t    = (const float*)d_in[2];
  const float* b_t    = (const float*)d_in[3];
  const float* gamma_ = (const float*)d_in[4];
  const float* beta_  = (const float*)d_in[5];
  // d_in[6..13]  = dec_weights[i]  (V_i x 768) f32
  // d_in[14..21] = dec_biases[i]   (V_i)       f32

  static const int VOC[NCLUST] = {3000, 4000, 5000, 6000, 3000, 4000, 5000, 6000};

  // Workspace layout (bytes)
  char* ws = (char*)d_ws;
  const size_t OFF_W   = 0;                                   // 768*768*2
  const size_t OFF_X   = OFF_W   + (size_t)HDIM * HDIM * 2;   // 8192*768*2
  const size_t OFF_H   = OFF_X   + (size_t)NTOK * HDIM * 2;
  const size_t OFF_DEC = OFF_H   + (size_t)NTOK * HDIM * 2;   // 36000*768*2
  const size_t OFF_CNT = OFF_DEC + (size_t)SUMV * HDIM * 2;
  const size_t OFF_LST = OFF_CNT + 256;                       // keep 256B aligned

  unsigned short* wbf   = (unsigned short*)(ws + OFF_W);
  unsigned short* xbf   = (unsigned short*)(ws + OFF_X);
  unsigned short* hbf   = (unsigned short*)(ws + OFF_H);
  unsigned short* decbf = (unsigned short*)(ws + OFF_DEC);
  int*            cnts  = (int*)(ws + OFF_CNT);
  int*            lists = (int*)(ws + OFF_LST);

  // 1) zero output (masked positions stay zero) and compaction counters
  hipMemsetAsync(d_out, 0, (size_t)out_size * sizeof(float), stream);
  hipMemsetAsync(cnts, 0, NCLUST * sizeof(int), stream);

  // 2) f32 -> bf16 conversions
  {
    int n = HDIM * HDIM;
    convert_bf16_kernel<<<(n + 255) / 256, 256, 0, stream>>>(W_t, wbf, n);
  }
  {
    int n = NTOK * HDIM;
    convert_bf16_kernel<<<(n + 255) / 256, 256, 0, stream>>>(x, xbf, n);
  }
  int wrow[NCLUST];
  long long ooff[NCLUST];
  {
    int rows = 0; long long oo = 0;
    for (int i = 0; i < NCLUST; ++i) {
      wrow[i] = rows;
      ooff[i] = oo;
      int n = VOC[i] * HDIM;
      convert_bf16_kernel<<<(n + 255) / 256, 256, 0, stream>>>(
          (const float*)d_in[6 + i], decbf + (size_t)rows * HDIM, n);
      rows += VOC[i];
      oo   += (long long)NTOK * VOC[i];
    }
  }

  // 3) token compaction per cluster
  compact_kernel<<<NTOK / 256, 256, 0, stream>>>(labels, cnts, lists, NTOK);

  // 4) fused transform (GEMM + bias + erf-GELU + LayerNorm) -> h bf16
  transform_kernel<<<NTOK / 16, 256, 0, stream>>>(xbf, wbf, b_t, gamma_, beta_, hbf);

  // 5) masked decoder GEMMs over compacted tokens
  DecParams p;
  p.h = hbf; p.wdec = decbf; p.counts = cnts; p.lists = lists;
  p.out = (float*)d_out;
  for (int i = 0; i < NCLUST; ++i) {
    p.biases[i]  = (const float*)d_in[14 + i];
    p.out_off[i] = ooff[i];
    p.vocab[i]   = VOC[i];
    p.wrow[i]    = wrow[i];
  }
  dim3 grid(12 /* ceil(6000/512) */, NTOK / 16, NCLUST);
  decoder_kernel<<<grid, 256, 0, stream>>>(p);
}
